// GNN_IF_33827162423530
// MI455X (gfx1250) — compile-verified, matrix operand order checked by reading, not run on hardware
//
#include <hip/hip_runtime.h>
#include <hip/hip_bf16.h>

#define NN 50000
#define NE 400000
#define DD 512
#define SELF_W 2.0f

typedef __attribute__((ext_vector_type(16))) __bf16 v16bf;
typedef __attribute__((ext_vector_type(8)))  float  v8f;

// pack two fp32 -> packed bf16x2 (round-to-nearest-even)
__device__ __forceinline__ unsigned int pk2bf(float a, float b) {
    unsigned int ua = __float_as_uint(a);
    unsigned int ub = __float_as_uint(b);
    ua += 0x7FFFu + ((ua >> 16) & 1u);
    ub += 0x7FFFu + ((ub >> 16) & 1u);
    return (ua >> 16) | (ub & 0xFFFF0000u);
}

// ======================= CSR build (by destination) =======================
__global__ void k_zeroi(int* __restrict__ p, int n) {
    int i = blockIdx.x * blockDim.x + threadIdx.x;
    if (i < n) p[i] = 0;
}

__global__ void k_counti(const int* __restrict__ dst, int* __restrict__ cnt, int e) {
    int i = blockIdx.x * blockDim.x + threadIdx.x;
    if (i < e) atomicAdd(&cnt[dst[i]], 1);
}

// per-256-block exclusive scan (Hillis-Steele in LDS) + block totals
__global__ __launch_bounds__(256)
void k_scan1(const int* __restrict__ cnt, int* __restrict__ excl,
             int* __restrict__ btot, int n) {
    __shared__ int s[256];
    int t = threadIdx.x, i = blockIdx.x * 256 + t;
    int v = (i < n) ? cnt[i] : 0;
    s[t] = v;
    __syncthreads();
#pragma unroll
    for (int off = 1; off < 256; off <<= 1) {
        int u = (t >= off) ? s[t - off] : 0;
        __syncthreads();
        s[t] += u;
        __syncthreads();
    }
    if (i < n) excl[i] = s[t] - v;           // exclusive within block
    if (t == 255) btot[blockIdx.x] = s[255]; // block total
}

// single-block exclusive scan of block totals
__global__ __launch_bounds__(256)
void k_scan2(const int* __restrict__ btot, int* __restrict__ boff, int nb) {
    __shared__ int s[256];
    int t = threadIdx.x;
    int v = (t < nb) ? btot[t] : 0;
    s[t] = v;
    __syncthreads();
#pragma unroll
    for (int off = 1; off < 256; off <<= 1) {
        int u = (t >= off) ? s[t - off] : 0;
        __syncthreads();
        s[t] += u;
        __syncthreads();
    }
    boff[t] = s[t] - v;
}

// combine scans -> row_start/cursor; also dinv = rsqrt(cnt + 2)
__global__ __launch_bounds__(256)
void k_scan3(const int* __restrict__ cnt, const int* __restrict__ excl,
             const int* __restrict__ boff, int* __restrict__ row_start,
             int* __restrict__ cursor, float* __restrict__ dinv, int n, int etot) {
    int i = blockIdx.x * 256 + threadIdx.x;
    if (i < n) {
        int rs = excl[i] + boff[blockIdx.x];
        row_start[i] = rs;
        cursor[i]    = rs;
        dinv[i]      = rsqrtf((float)cnt[i] + SELF_W);
    }
    if (i == 0) row_start[n] = etot;
}

__global__ void k_fill(const int* __restrict__ src, const int* __restrict__ dst,
                       int* __restrict__ cursor, int* __restrict__ ssrc, int e) {
    int i = blockIdx.x * blockDim.x + threadIdx.x;
    if (i < e) {
        int pos = atomicAdd(&cursor[dst[i]], 1);
        ssrc[pos] = src[i];
    }
}

// ======================= W -> bf16 transposed =======================
// Wt[n][k] = bf16(W[k][n]); 512x512; each thread emits 8 contiguous k's (one uint4)
__global__ __launch_bounds__(256)
void k_wt_bf16(const float* __restrict__ W, unsigned short* __restrict__ Wt) {
    int i = blockIdx.x * blockDim.x + threadIdx.x;   // 0 .. 32767
    int n  = i & 511;
    int k8 = (i >> 9) << 3;
    uint4 u;
    u.x = pk2bf(W[(size_t)(k8 + 0) * DD + n], W[(size_t)(k8 + 1) * DD + n]);
    u.y = pk2bf(W[(size_t)(k8 + 2) * DD + n], W[(size_t)(k8 + 3) * DD + n]);
    u.z = pk2bf(W[(size_t)(k8 + 4) * DD + n], W[(size_t)(k8 + 5) * DD + n]);
    u.w = pk2bf(W[(size_t)(k8 + 6) * DD + n], W[(size_t)(k8 + 7) * DD + n]);
    *(uint4*)(Wt + (size_t)n * DD + k8) = u;
}

// ======================= GEMM: H = X @ W (bf16 WMMA, f32 acc) =======================
// 256 threads = 8 waves; block tile 128(M) x 64(N); K-step 32.
__global__ __launch_bounds__(256)
void k_gemm_bf16(const float* __restrict__ X, const unsigned short* __restrict__ Wt,
                 float* __restrict__ H, int M) {
    __shared__ unsigned short sA[128][40];
    __shared__ unsigned short sB[64][40];

    const int tid  = threadIdx.x;
    const int lane = tid & 31;
    const int wave = tid >> 5;
    const int l15  = lane & 15;
    const int kh   = (lane >> 4) << 3;
    const int m0   = blockIdx.y * 128;
    const int n0   = blockIdx.x * 64;
    const int arow = wave * 16 + l15;

    const int bn = tid >> 2;
    const int bk = (tid & 3) << 3;
    const unsigned short* WtRow = Wt + (size_t)(n0 + bn) * DD + bk;

    v8f acc[4];
#pragma unroll
    for (int t = 0; t < 4; ++t) acc[t] = (v8f)0.f;

    for (int k0 = 0; k0 < DD; k0 += 32) {
        __syncthreads();
#pragma unroll
        for (int i = 0; i < 2; ++i) {
            int idx = tid + (i << 8);
            int r = idx >> 2, kq = (idx & 3) << 3;
            float4 a0 = make_float4(0.f, 0.f, 0.f, 0.f), a1 = a0;
            int gr = m0 + r;
            if (gr < M) {
                const float* p = X + (size_t)gr * DD + k0 + kq;
                a0 = *(const float4*)p;
                a1 = *(const float4*)(p + 4);
            }
            uint4 u;
            u.x = pk2bf(a0.x, a0.y); u.y = pk2bf(a0.z, a0.w);
            u.z = pk2bf(a1.x, a1.y); u.w = pk2bf(a1.z, a1.w);
            *(uint4*)&sA[r][kq] = u;
        }
        *(uint4*)&sB[bn][bk] = *(const uint4*)(WtRow + k0);
        __syncthreads();

        v16bf a;
        ((int4*)&a)[0] = *(const int4*)&sA[arow][kh];
        ((int4*)&a)[1] = *(const int4*)&sA[arow][kh + 16];
        v16bf bfr[4];
#pragma unroll
        for (int t = 0; t < 4; ++t) {
            const int brow = t * 16 + l15;
            ((int4*)&bfr[t])[0] = *(const int4*)&sB[brow][kh];
            ((int4*)&bfr[t])[1] = *(const int4*)&sB[brow][kh + 16];
        }
#pragma unroll
        for (int t = 0; t < 4; ++t) {
            acc[t] = __builtin_amdgcn_wmma_f32_16x16x32_bf16(
                false, a, false, bfr[t], (short)0, acc[t], false, false);
        }
    }

    const int mbase = m0 + wave * 16 + ((lane >> 4) << 3);
#pragma unroll
    for (int t = 0; t < 4; ++t) {
        int col = n0 + t * 16 + l15;
#pragma unroll
        for (int r = 0; r < 8; ++r) {
            int row = mbase + r;
            if (row < M) H[(size_t)row * DD + col] = acc[t][r];
        }
    }
}

// ======================= fused gather aggregation =======================
// One block per destination node; 128 threads x float4 = 512 features.
// out[d] = relu( dinv[d] * sum_{e in in(d)} dinv[src]*H[src]
//                + 2*dinv[d]^2 * H[d] + bias )
__global__ __launch_bounds__(128)
void k_gather(const int* __restrict__ row_start, const int* __restrict__ ssrc,
              const float* __restrict__ dinv, const float* __restrict__ H,
              const float* __restrict__ bias, float* __restrict__ out) {
    const int d  = blockIdx.x;
    const int c  = threadIdx.x << 2;
    const int j0 = row_start[d];
    const int j1 = row_start[d + 1];
    const float* Hc = H + c;

    float4 acc = make_float4(0.f, 0.f, 0.f, 0.f);
    for (int j = j0; j < j1; ++j) {
        int s = ssrc[j];
        float w = dinv[s];
        float4 h = *(const float4*)(Hc + (size_t)s * DD);
        acc.x = fmaf(h.x, w, acc.x);
        acc.y = fmaf(h.y, w, acc.y);
        acc.z = fmaf(h.z, w, acc.z);
        acc.w = fmaf(h.w, w, acc.w);
    }
    float dd  = dinv[d];
    float slf = SELF_W * dd * dd;
    float4 hd = *(const float4*)(Hc + (size_t)d * DD);
    float4 b  = *(const float4*)(bias + c);
    float4 r;
    r.x = fmaxf(fmaf(dd, acc.x, fmaf(slf, hd.x, b.x)), 0.f);
    r.y = fmaxf(fmaf(dd, acc.y, fmaf(slf, hd.y, b.y)), 0.f);
    r.z = fmaxf(fmaf(dd, acc.z, fmaf(slf, hd.z, b.z)), 0.f);
    r.w = fmaxf(fmaf(dd, acc.w, fmaf(slf, hd.w, b.w)), 0.f);
    *(float4*)(out + (size_t)d * DD + c) = r;
}

// ======================= launch =======================
extern "C" void kernel_launch(void* const* d_in, const int* in_sizes, int n_in,
                              void* d_out, int out_size, void* d_ws, size_t ws_size,
                              hipStream_t stream) {
    (void)in_sizes; (void)n_in; (void)out_size; (void)ws_size;
    const float* X  = (const float*)d_in[0];
    const int*   ei = (const int*)d_in[1];
    const float* W1 = (const float*)d_in[2];
    const float* b1 = (const float*)d_in[3];
    const float* W2 = (const float*)d_in[4];
    const float* b2 = (const float*)d_in[5];
    float* out = (float*)d_out;

    const int* src = ei;
    const int* dst = ei + NE;

    // ---- workspace layout (~105.6 MB) ----
    char* w = (char*)d_ws;
    float* dinv        = (float*)w;           w += 51200 * 4;
    float* H           = (float*)w;           w += (size_t)NN * DD * 4;
    unsigned short* Wt = (unsigned short*)w;  w += (size_t)DD * DD * 2;
    int* cnt           = (int*)w;             w += 50176 * 4;
    int* excl          = (int*)w;             w += 50176 * 4;
    int* row_start     = (int*)w;             w += 50176 * 4;  // uses NN+1
    int* cursor        = (int*)w;             w += 50176 * 4;
    int* btot          = (int*)w;             w += 256 * 4;
    int* boff          = (int*)w;             w += 256 * 4;
    int* ssrc          = (int*)w;             w += (size_t)NE * 4;

    const int nbScan = (NN + 255) / 256;           // 196
    dim3 gGemm(DD / 64, (NN + 127) / 128);         // 8 x 391

    // ---- CSR build + normalization (shared by both layers) ----
    k_zeroi <<<nbScan, 256, 0, stream>>>(cnt, NN);
    k_counti<<<(NE + 255) / 256, 256, 0, stream>>>(dst, cnt, NE);
    k_scan1 <<<nbScan, 256, 0, stream>>>(cnt, excl, btot, NN);
    k_scan2 <<<1, 256, 0, stream>>>(btot, boff, nbScan);
    k_scan3 <<<nbScan, 256, 0, stream>>>(cnt, excl, boff, row_start, cursor, dinv, NN, NE);
    k_fill  <<<(NE + 255) / 256, 256, 0, stream>>>(src, dst, cursor, ssrc, NE);

    // ---- layer 1 ----
    k_wt_bf16  <<<128, 256, 0, stream>>>(W1, Wt);
    k_gemm_bf16<<<gGemm, 256, 0, stream>>>(X, Wt, H, NN);
    k_gather   <<<NN, 128, 0, stream>>>(row_start, ssrc, dinv, H, b1, out);
    // d_out holds X1 = relu(layer1)

    // ---- layer 2 ----
    k_wt_bf16  <<<128, 256, 0, stream>>>(W2, Wt);
    k_gemm_bf16<<<gGemm, 256, 0, stream>>>(out, Wt, H, NN);
    k_gather   <<<NN, 128, 0, stream>>>(row_start, ssrc, dinv, H, b2, out);
}
